// VariableNodesAttentionLayer_35029753266783
// MI455X (gfx1250) — matrix operand order repaired and checked by reading, not compiled, not run on hardware
//
#include <hip/hip_runtime.h>
#include <math.h>

// ---------------------------------------------------------------------------
// Fused learned-query cross-attention for MI455X (gfx1250, wave32, WMMA).
// bf16 matrix cores with f32 accumulation; flash-attention style online
// softmax, fully fused projections (a = xWa, v = xWv + bv stay in LDS).
// Head-pair split across workgroups keeps per-wave state spill-free.
// ---------------------------------------------------------------------------

#define HEADS   4
#define ADIMS   64
#define VDIMS   64
#define NODESN  128
#define FD      256
#define NBATCH  8
#define NTOK    8192
#define NCH     32          // token chunks (split-softmax partials)
#define CHUNK   256         // tokens per workgroup
#define SUBT    64          // tokens per inner iteration
#define LN_EPS  1e-5f

typedef __attribute__((ext_vector_type(16))) __bf16 v16bf;
typedef __attribute__((ext_vector_type(8)))  __bf16 v8bf;
typedef __attribute__((ext_vector_type(8)))  float  v8f;

union FragU { v16bf v; v8bf h[2]; };

static __device__ __forceinline__ unsigned pkbf(__bf16 a, __bf16 b) {
  unsigned short xa, xb;
  __builtin_memcpy(&xa, &a, 2);
  __builtin_memcpy(&xb, &b, 2);
  return (unsigned)xa | ((unsigned)xb << 16);
}

static __device__ __forceinline__ v8f wmma_bf16(v16bf a, v16bf b, v8f c) {
  // v_wmma_f32_16x16x32_bf16: D = A(16x32) x B(32x16) + C(16x16 f32)
  return __builtin_amdgcn_wmma_f32_16x16x32_bf16(
      /*neg_a=*/false, a, /*neg_b=*/false, b,
      /*c_mod=*/(short)0, c, /*reuse_a=*/false, /*reuse_b=*/false);
}

// A-operand 16x32 bf16 fragment from row-major storage [M][K] (K contiguous).
// Layout used here (contiguous k, not ISA table order): lane m=lane&15 holds row M=m; half=lane>>4 selects
// K 0..15 (half 0) or 16..31 (half 1), contiguous (matches B fragments).
static __device__ __forceinline__ v16bf load_a_frag(const __bf16* base,
                                                    int stride, int lane) {
  const int m = lane & 15, half = lane >> 4;
  const __bf16* p = base + m * stride + half * 16;
  FragU f;
  f.h[0] = *(const v8bf*)(p);
  f.h[1] = *(const v8bf*)(p + 8);
  return f.v;
}

// B-operand 32x16 bf16 fragment from N-major storage [N][K] (K contiguous).
// Layout used here (contiguous k, not ISA table order): lane n=lane&15 holds column N=n; lanes 0-15 K=0..15,
// lanes 16-31 K=16..31 (2 packed K per VGPR).
static __device__ __forceinline__ v16bf load_b_frag(const __bf16* base,
                                                    int stride, int lane) {
  const int n = lane & 15, half = lane >> 4;
  const __bf16* p = base + n * stride + half * 16;
  FragU f;
  f.h[0] = *(const v8bf*)(p);
  f.h[1] = *(const v8bf*)(p + 8);
  return f.v;
}

// ---------------------------------------------------------------------------
// Prep: permute Wa/Wv (256x256 f32, row = K, col = N) into exact B-fragment
// order so the GEMM loads are two contiguous global_load_b128 per lane.
// Fragment index: ((kt*16 + ct)*32 + lane)*16 + e ;
//   lane<16: k = kt*32 + e,       n = ct*16 + lane
//   lane>=16: k = kt*32 + 16 + e, n = ct*16 + lane-16
// ---------------------------------------------------------------------------
__global__ void prep_weights_kernel(const float* __restrict__ Wa,
                                    const float* __restrict__ Wv,
                                    __bf16* __restrict__ WBa,
                                    __bf16* __restrict__ WBv) {
  int idx = blockIdx.x * blockDim.x + threadIdx.x;   // 0..131071
  int sel = idx >> 16;
  int i   = idx & 65535;
  int e    = i & 15;
  int lane = (i >> 4) & 31;
  int fc   = i >> 9;                                  // 0..127
  int ct = fc & 15, kt = fc >> 4;
  int k = kt * 32 + ((lane >> 4) << 4) + e;
  int n = ct * 16 + (lane & 15);
  const float* W = sel ? Wv : Wa;
  __bf16*      D = sel ? WBv : WBa;
  D[i] = (__bf16)W[k * FD + n];
}

// nodes -> bf16, pre-scaled by A_DIMS^-0.5 (scaling A pre-GEMM is exact)
__global__ void prep_nodes_kernel(const float* __restrict__ nodes,
                                  __bf16* __restrict__ nodesbf) {
  int i = blockIdx.x * blockDim.x + threadIdx.x;      // 0..8191
  nodesbf[i] = (__bf16)(nodes[i] * 0.125f);
}

// ---------------------------------------------------------------------------
// Fused projection + flash attention, one head-pair per workgroup.
// grid.x = NBATCH * NCH * 2 ; block = 512 threads (16 wave32), occupancy 1
// so each wave may use the full 256-VGPR budget (4 waves/SIMD).
// Projection: waves 0..7 -> 'a' col-tile ct=w, waves 8..15 -> 'v' ct=w-8.
// Attention:  wave w -> node-tile nt=w&7, local head hl=w>>3 (one task).
// ---------------------------------------------------------------------------
__global__ void __launch_bounds__(512, 1)
attn_fused_kernel(const float*  __restrict__ x,
                  const float*  __restrict__ bv,
                  const __bf16* __restrict__ WBa,
                  const __bf16* __restrict__ WBv,
                  const __bf16* __restrict__ nodesbf,
                  float* __restrict__ partm,
                  float* __restrict__ partl,
                  float* __restrict__ partacc) {
  extern __shared__ __bf16 smem[];
  __bf16* xs  = smem;                   // 64 x 264   staged x (bf16)
  __bf16* as_ = xs  + 64 * 264;         // 64 x 136   a cols for 2 heads
  __bf16* vts = as_ + 64 * 136;         // 128 x 72   v (dim-major) 2 heads
  __bf16* ps  = vts + 128 * 72;         // 16 waves x (16 x 72)  P staging

  const int tid  = threadIdx.x;
  const int lane = tid & 31;
  const int w    = tid >> 5;            // 0..15
  const int half = lane >> 4;
  const int ln   = lane & 15;

  const int bx   = blockIdx.x;          // ((b*NCH + cch)*2 + hp)
  const int hp   = bx & 1;              // head pair: heads hp*2, hp*2+1
  const int cch  = (bx >> 1) & 31;      // chunk
  const int b    = bx >> 6;             // batch
  const int tok0 = cch * CHUNK;

  // ---- projection role (uniform per wave): 8 local col-tiles each side
  const bool isV = (w >= 8);
  const int  wl  = w & 7;               // local column tile (16 cols)
  const __bf16* WB = isV ? WBv : WBa;
  const int colN   = wl * 16 + ln;              // local col 0..127
  const int ctGlob = hp * 8 + wl;               // global col tile 0..15
  const float bvv = isV ? bv[hp * 128 + colN] : 0.f;

  // ---- attention task
  const int nt = w & 7;                 // node tile (16 nodes)
  const int hl = w >> 3;                // local head (0/1)
  const int h  = hp * 2 + hl;           // global head

  v8f  acc[4];
  float mrow[8], lrow[8];
#pragma unroll
  for (int d = 0; d < 4; ++d)
#pragma unroll
    for (int r = 0; r < 8; ++r) acc[d][r] = 0.f;
#pragma unroll
  for (int r = 0; r < 8; ++r) { mrow[r] = -3.0e38f; lrow[r] = 0.f; }

  for (int s = 0; s < CHUNK / SUBT; ++s) {
    const int jb = tok0 + s * SUBT;

    // ---- stage 64x256 f32 of x into LDS as bf16 (coalesced float4 loads)
#pragma unroll
    for (int q = 0; q < 8; ++q) {
      int fidx = q * 512 + tid;               // float4 index, 4096 total
      int row = fidx >> 6, c4 = fidx & 63;
      const float4 xv =
          *(const float4*)(x + ((long)(b * NTOK + jb + row)) * FD + c4 * 4);
      uint2 u;
      u.x = pkbf((__bf16)xv.x, (__bf16)xv.y);
      u.y = pkbf((__bf16)xv.z, (__bf16)xv.w);
      *(uint2*)(xs + row * 264 + c4 * 4) = u;
    }
    __syncthreads();

    // ---- projection: one col-tile per wave, 4 row-tiles, K=256.
    // One B-fragment per k-step serves all 4 row tiles; 4 independent
    // accumulator chains interleave to hide WMMA->WMMA hazards.
    {
      v8f cacc[4];
#pragma unroll
      for (int rt = 0; rt < 4; ++rt)
#pragma unroll
        for (int r = 0; r < 8; ++r) cacc[rt][r] = 0.f;
#pragma unroll
      for (int kt = 0; kt < 8; ++kt) {
        const __bf16* wb = WB + (((kt * 16 + ctGlob) * 32 + lane) << 4);
        FragU bf_;
        bf_.h[0] = *(const v8bf*)(wb);
        bf_.h[1] = *(const v8bf*)(wb + 8);
#pragma unroll
        for (int rt = 0; rt < 4; ++rt) {
          v16bf af = load_a_frag(xs + (rt * 16) * 264 + kt * 32, 264, lane);
          cacc[rt] = wmma_bf16(af, bf_.v, cacc[rt]);
        }
      }
#pragma unroll
      for (int rt = 0; rt < 4; ++rt) {
        const int rowb = rt * 16 + half * 8;   // token row base
        if (isV) {
          __bf16 e[8];
#pragma unroll
          for (int r = 0; r < 8; ++r) e[r] = (__bf16)(cacc[rt][r] + bvv);
          uint4 u;
          u.x = pkbf(e[0], e[1]); u.y = pkbf(e[2], e[3]);
          u.z = pkbf(e[4], e[5]); u.w = pkbf(e[6], e[7]);
          *(uint4*)(vts + colN * 72 + rowb) = u;   // dim-major (transposed)
        } else {
#pragma unroll
          for (int r = 0; r < 8; ++r)
            as_[(rowb + r) * 136 + colN] = (__bf16)cacc[rt][r];
        }
      }
    }
    __syncthreads();

    // ---- attention: one (node-tile, head) task per wave, online softmax.
    // node A-fragments reloaded per sub-chunk (L2 resident) to keep them
    // out of the persistent register set.
    {
      const v16bf naf0 =
          load_a_frag(nodesbf + (nt * 16) * ADIMS + 0, ADIMS, lane);
      const v16bf naf1 =
          load_a_frag(nodesbf + (nt * 16) * ADIMS + 32, ADIMS, lane);
      // S = nodes x a^T : 4 token tiles, K=64 (2 wmma each); nodes were
      // pre-scaled by A_DIMS^-0.5 so S is already scaled.
      v8f st[4];
#pragma unroll
      for (int tt = 0; tt < 4; ++tt) {
        v8f cs;
#pragma unroll
        for (int r = 0; r < 8; ++r) cs[r] = 0.f;
        v16bf b0 = load_b_frag(as_ + (tt * 16) * 136 + hl * 64 + 0, 136, lane);
        cs = wmma_bf16(naf0, b0, cs);
        v16bf b1 = load_b_frag(as_ + (tt * 16) * 136 + hl * 64 + 32, 136, lane);
        cs = wmma_bf16(naf1, b1, cs);
        st[tt] = cs;
      }
      // row max (rows = nodes live across a 16-lane group)
      float mnew[8], alpha[8];
#pragma unroll
      for (int r = 0; r < 8; ++r) {
        float mx = fmaxf(fmaxf(st[0][r], st[1][r]), fmaxf(st[2][r], st[3][r]));
        mx = fmaxf(mx, __shfl_xor(mx, 1, 32));
        mx = fmaxf(mx, __shfl_xor(mx, 2, 32));
        mx = fmaxf(mx, __shfl_xor(mx, 4, 32));
        mx = fmaxf(mx, __shfl_xor(mx, 8, 32));
        mnew[r]  = fmaxf(mrow[r], mx);
        alpha[r] = __expf(mrow[r] - mnew[r]);
        mrow[r]  = mnew[r];
      }
      // P = exp(S - m); stage to wave-private LDS (C-layout -> A-layout)
      __bf16* psw = ps + w * 16 * 72;
#pragma unroll
      for (int tt = 0; tt < 4; ++tt) {
#pragma unroll
        for (int r = 0; r < 8; ++r) {
          float pv = __expf(st[tt][r] - mnew[r]);
          st[tt][r] = pv;
          psw[(half * 8 + r) * 72 + tt * 16 + ln] = (__bf16)pv;
        }
      }
#pragma unroll
      for (int r = 0; r < 8; ++r) {
        float sm = st[0][r] + st[1][r] + st[2][r] + st[3][r];
        sm += __shfl_xor(sm, 1, 32);
        sm += __shfl_xor(sm, 2, 32);
        sm += __shfl_xor(sm, 4, 32);
        sm += __shfl_xor(sm, 8, 32);
        lrow[r] = lrow[r] * alpha[r] + sm;
#pragma unroll
        for (int d = 0; d < 4; ++d) acc[d][r] *= alpha[r];
      }
      // acc += P x V : K = 64 tokens (2 k-steps) x 4 dim tiles
      v16bf pa0 = load_a_frag(psw + 0,  72, lane);
      v16bf pa1 = load_a_frag(psw + 32, 72, lane);
#pragma unroll
      for (int d = 0; d < 4; ++d) {
        v16bf vb0 = load_b_frag(vts + (hl * 64 + d * 16) * 72 + 0,  72, lane);
        acc[d] = wmma_bf16(pa0, vb0, acc[d]);
        v16bf vb1 = load_b_frag(vts + (hl * 64 + d * 16) * 72 + 32, 72, lane);
        acc[d] = wmma_bf16(pa1, vb1, acc[d]);
      }
    }
    __syncthreads();
  }

  // ---- write split-softmax partials
  {
    const long base = (((long)b * NCH + cch) * HEADS + h) * NODESN + nt * 16;
    if (ln == 0) {
#pragma unroll
      for (int r = 0; r < 8; ++r) {
        partm[base + half * 8 + r] = mrow[r];
        partl[base + half * 8 + r] = lrow[r];
      }
    }
#pragma unroll
    for (int d = 0; d < 4; ++d)
#pragma unroll
      for (int r = 0; r < 8; ++r)
        partacc[(base + half * 8 + r) * 64 + d * 16 + ln] = acc[d][r];
  }
}

// ---------------------------------------------------------------------------
// Combine chunk partials, leaky-ReLU, LayerNorm over 256 dims.
// grid.x = NBATCH*NODESN ; 256 threads (thread d: head d>>6, dim d&63).
// ---------------------------------------------------------------------------
__global__ void __launch_bounds__(256)
reduce_ln_kernel(const float* __restrict__ partm,
                 const float* __restrict__ partl,
                 const float* __restrict__ partacc,
                 const float* __restrict__ gamma,
                 const float* __restrict__ beta,
                 float* __restrict__ out) {
  const int blk = blockIdx.x;
  const int b = blk >> 7, node = blk & 127;
  const int d = threadIdx.x, h = d >> 6, dd = d & 63;

  float mc[NCH];
  float M = -3.0e38f;
#pragma unroll
  for (int c = 0; c < NCH; ++c) {
    mc[c] = partm[(((long)b * NCH + c) * HEADS + h) * NODESN + node];
    M = fmaxf(M, mc[c]);
  }
  float o = 0.f, L = 0.f;
#pragma unroll
  for (int c = 0; c < NCH; ++c) {
    const long i = (((long)b * NCH + c) * HEADS + h) * NODESN + node;
    const float wgt = __expf(mc[c] - M);
    L += wgt * partl[i];
    o += wgt * partacc[i * 64 + dd];
  }
  float val = o / L;
  val = val >= 0.f ? val : 0.2f * val;            // leaky_relu(0.2)

  // block LayerNorm over 256 values
  __shared__ float redS[8], redQ[8];
  float s = val, q = val * val;
#pragma unroll
  for (int off = 1; off < 32; off <<= 1) {
    s += __shfl_xor(s, off, 32);
    q += __shfl_xor(q, off, 32);
  }
  const int wv = threadIdx.x >> 5;
  if ((threadIdx.x & 31) == 0) { redS[wv] = s; redQ[wv] = q; }
  __syncthreads();
  float S = 0.f, Q = 0.f;
#pragma unroll
  for (int i = 0; i < 8; ++i) { S += redS[i]; Q += redQ[i]; }
  const float mu  = S * (1.f / 256.f);
  const float var = Q * (1.f / 256.f) - mu * mu;
  const float rs  = rsqrtf(var + LN_EPS);
  out[(long)blk * 256 + d] = (val - mu) * rs * gamma[d] + beta[d];
}

// ---------------------------------------------------------------------------
extern "C" void kernel_launch(void* const* d_in, const int* in_sizes, int n_in,
                              void* d_out, int out_size, void* d_ws,
                              size_t ws_size, hipStream_t stream) {
  (void)in_sizes; (void)n_in; (void)out_size; (void)ws_size;
  const float* x     = (const float*)d_in[0];
  const float* nodes = (const float*)d_in[1];
  const float* Wa    = (const float*)d_in[2];
  const float* Wv    = (const float*)d_in[3];
  const float* bv    = (const float*)d_in[4];
  const float* gamma = (const float*)d_in[5];
  const float* beta  = (const float*)d_in[6];

  char* ws = (char*)d_ws;
  __bf16* WBa     = (__bf16*)(ws);                       // 128 KiB
  __bf16* WBv     = (__bf16*)(ws + 131072);              // 128 KiB
  __bf16* nodesbf = (__bf16*)(ws + 262144);              // 16 KiB
  float*  partm   = (float*)(ws + 278528);               // 512 KiB
  float*  partl   = (float*)(ws + 278528 + 524288);      // 512 KiB
  float*  partacc = (float*)(ws + 278528 + 2 * 524288);  // 32 MiB

  prep_weights_kernel<<<512, 256, 0, stream>>>(Wa, Wv, WBa, WBv);
  prep_nodes_kernel<<<32, 256, 0, stream>>>(nodes, nodesbf);

  const size_t smem =
      (size_t)(64 * 264 + 64 * 136 + 128 * 72 + 16 * 16 * 72) *
      sizeof(__bf16);
  attn_fused_kernel<<<NBATCH * NCH * 2, 512, smem, stream>>>(
      x, bv, WBa, WBv, nodesbf, partm, partl, partacc);

  reduce_ln_kernel<<<NBATCH * NODESN, 256, 0, stream>>>(
      partm, partl, partacc, gamma, beta, (float*)d_out);
}
